// Mamba_69784628625647
// MI455X (gfx1250) — matrix operand order repaired and checked by reading, not compiled, not run on hardware
//
#include <hip/hip_runtime.h>
#include <hip/hip_bf16.h>
#include <math.h>

// ---------------- problem constants ----------------
#define D_MODEL 1024
#define ED      2048
#define NSTATE  16
#define DT_RANK 64
#define KCONV   4
#define B_SZ    4
#define LSEQ    512
#define BL      (B_SZ * LSEQ)   // 2048 rows of (b,l)

// ---------------- vector types ----------------
typedef __attribute__((ext_vector_type(16))) _Float16     v16h;
typedef __attribute__((ext_vector_type(8)))  _Float16     h8;
typedef __attribute__((ext_vector_type(8)))  float        v8f;
typedef __attribute__((ext_vector_type(4)))  float        f4;
typedef __attribute__((ext_vector_type(4)))  unsigned int u32x4;
typedef __attribute__((ext_vector_type(8)))  int          i32x8;
typedef __attribute__((ext_vector_type(4)))  int          i32x4;

union AFrag { v16h v; h8 h[2]; };

#if __has_builtin(__builtin_amdgcn_tensor_load_to_lds)
#define HAVE_TDM 1
#else
#define HAVE_TDM 0
#endif

#if HAVE_TDM
// 2D tile load Global->LDS via Tensor Data Mover.
// D# per ISA cdna5 §8.3/8.4: group0 = {count, lds_addr, global_addr, type=2},
// group1 = {data_size=2B, tensor_dim0=K, tensor_dim1=rows, tile 64x64, stride=K}.
__device__ __forceinline__ void tdm_load_2d_f16(unsigned lds_off,
                                                const _Float16* gbase,
                                                unsigned rows_total,
                                                unsigned K,
                                                unsigned tile_k,
                                                unsigned tile_rows)
{
  unsigned long long ga = (unsigned long long)(size_t)gbase;
  u32x4 g0;
  g0[0] = 1u;                                              // count=1, user mode
  g0[1] = lds_off;                                         // lds_addr (bytes)
  g0[2] = (unsigned)(ga & 0xFFFFFFFFu);                    // global_addr[31:0]
  g0[3] = (unsigned)((ga >> 32) & 0x01FFFFFFu) | (2u << 30); // addr[56:32] | type=2
  i32x8 g1;
  g1[0] = (int)(1u << 16);                                 // data_size=1 -> 2 bytes
  g1[1] = (int)((K & 0xFFFFu) << 16);                      // tensor_dim0 lo16
  g1[2] = (int)(((K >> 16) & 0xFFFFu) | ((rows_total & 0xFFFFu) << 16)); // dim0 hi | dim1 lo
  g1[3] = (int)(((rows_total >> 16) & 0xFFFFu) | ((tile_k & 0xFFFFu) << 16)); // dim1 hi | tile_dim0
  g1[4] = (int)(tile_rows & 0xFFFFu);                      // tile_dim1 (tile_dim2=0)
  g1[5] = (int)K;                                          // tensor_dim0_stride lo32
  g1[6] = 0;
  g1[7] = 0;
  i32x4 z4 = {0, 0, 0, 0};
#if defined(__clang_major__) && (__clang_major__ >= 23)
  i32x8 z8 = {0, 0, 0, 0, 0, 0, 0, 0};
  __builtin_amdgcn_tensor_load_to_lds(g0, g1, z4, z4, z8, 0);
#else
  __builtin_amdgcn_tensor_load_to_lds(g0, g1, z4, z4, 0);
#endif
}
#endif

// ---------------- f32 -> f16 conversion ----------------
__global__ void f32_to_f16_kernel(const float* __restrict__ src,
                                  _Float16* __restrict__ dst, int n) {
  int i = blockIdx.x * blockDim.x + threadIdx.x;
  if (i < n) dst[i] = (_Float16)src[i];
}

// ------------- WMMA GEMM: C[M,N] = A[M,K] * B[N,K]^T -------------
// One wave per 16 x (16*NT) strip: A fragment loaded once per k-step and fed
// to NT back-to-back v_wmma ops (A traffic / NT, independent WMMAs for the
// matrix pipe). DIRS=2: blockIdx.z selects direction (A/C dir-strided,
// per-dir weight/bias pointers).  MODE 1: softplus(acc + bias[n]) epilogue.
template <int NT, int MODE, int DIRS>
__global__ void wmma_gemm_nt_kernel(const _Float16* __restrict__ A,
                                    const _Float16* __restrict__ B0,
                                    const _Float16* __restrict__ B1,
                                    float* __restrict__ C,
                                    int M, int N, int K,
                                    const float* __restrict__ bias0,
                                    const float* __restrict__ bias1)
{
  const int lane = threadIdx.x & 31;
  const int grp  = lane >> 4;
  const int idx  = lane & 15;
  const int dir  = (DIRS == 2) ? (int)blockIdx.z : 0;

  const _Float16* Bw   = (DIRS == 2 && dir) ? B1 : B0;
  const float*    bias = (DIRS == 2 && dir) ? bias1 : bias0;
  const _Float16* Ad   = A + (size_t)dir * M * K;
  float*          Cd   = C + (size_t)dir * M * N;

  const int m0 = blockIdx.x * 16;
  const int n0 = blockIdx.y * (16 * NT);

  const _Float16* arow = Ad + (size_t)(m0 + idx) * K;
  const _Float16* brow[NT];
#pragma unroll
  for (int t = 0; t < NT; ++t)
    brow[t] = Bw + (size_t)(n0 + 16 * t + idx) * K;

  v8f acc[NT] = {};
  for (int k0 = 0; k0 < K; k0 += 32) {
    if (k0 + 32 < K) {
      __builtin_prefetch((const void*)(arow + k0 + 32), 0, 1);
#pragma unroll
      for (int t = 0; t < NT; ++t)
        __builtin_prefetch((const void*)(brow[t] + k0 + 32), 0, 1);
    }
    // A fragment (16x32 f16): lanes 0-15: K=k0+[0..7], k0+16+[0..7];
    // lanes 16-31: +8.  B fragment: same K order as A, n = lane&15.
    AFrag a;
    a.h[0] = *(const h8*)(arow + k0 + grp * 8);
    a.h[1] = *(const h8*)(arow + k0 + 16 + grp * 8);
#pragma unroll
    for (int t = 0; t < NT; ++t) {
      AFrag b;
      b.h[0] = *(const h8*)(brow[t] + k0 + grp * 8);
      b.h[1] = *(const h8*)(brow[t] + k0 + 16 + grp * 8);
      acc[t] = __builtin_amdgcn_wmma_f32_16x16x32_f16(
          false, a.v, false, b.v, (short)0, acc[t], false, false);
    }
  }

#pragma unroll
  for (int t = 0; t < NT; ++t) {
    const int n = n0 + 16 * t + idx;
    float bn = (MODE == 1) ? bias[n] : 0.f;
#pragma unroll
    for (int r = 0; r < 8; ++r) {
      float v = acc[t][r];
      if (MODE == 1) {
        v += bn;
        v = (v > 20.f) ? v : log1pf(__expf(v));   // softplus
      }
      Cd[(size_t)(m0 + r + grp * 8) * N + n] = v;
    }
  }
}

// Variant with f32 A (row stride lda), cvt to f16 in-register.
// delta = softplus(dbc[:, :64] @ Wdt^T + b_dt), K=64.
template <int NT, int DIRS>
__global__ void wmma_gemm_nt_a32_kernel(const float* __restrict__ A,
                                        const _Float16* __restrict__ B0,
                                        const _Float16* __restrict__ B1,
                                        float* __restrict__ C,
                                        int M, int N, int K, int lda,
                                        const float* __restrict__ bias0,
                                        const float* __restrict__ bias1)
{
  const int lane = threadIdx.x & 31;
  const int grp  = lane >> 4;
  const int idx  = lane & 15;
  const int dir  = (DIRS == 2) ? (int)blockIdx.z : 0;

  const _Float16* Bw   = (DIRS == 2 && dir) ? B1 : B0;
  const float*    bias = (DIRS == 2 && dir) ? bias1 : bias0;
  const float*    Ad   = A + (size_t)dir * M * lda;
  float*          Cd   = C + (size_t)dir * M * N;

  const int m0 = blockIdx.x * 16;
  const int n0 = blockIdx.y * (16 * NT);

  const float* arow = Ad + (size_t)(m0 + idx) * lda;
  const _Float16* brow[NT];
#pragma unroll
  for (int t = 0; t < NT; ++t)
    brow[t] = Bw + (size_t)(n0 + 16 * t + idx) * K;

  v8f acc[NT] = {};
  for (int k0 = 0; k0 < K; k0 += 32) {
    f4 a0 = *(const f4*)(arow + k0 + grp * 8);
    f4 a1 = *(const f4*)(arow + k0 + grp * 8 + 4);
    f4 a2 = *(const f4*)(arow + k0 + 16 + grp * 8);
    f4 a3 = *(const f4*)(arow + k0 + 16 + grp * 8 + 4);
    AFrag a;
#pragma unroll
    for (int i = 0; i < 4; ++i) {
      a.v[i]      = (_Float16)a0[i];
      a.v[4 + i]  = (_Float16)a1[i];
      a.v[8 + i]  = (_Float16)a2[i];
      a.v[12 + i] = (_Float16)a3[i];
    }
#pragma unroll
    for (int t = 0; t < NT; ++t) {
      AFrag b;
      b.h[0] = *(const h8*)(brow[t] + k0 + grp * 8);
      b.h[1] = *(const h8*)(brow[t] + k0 + 16 + grp * 8);
      acc[t] = __builtin_amdgcn_wmma_f32_16x16x32_f16(
          false, a.v, false, b.v, (short)0, acc[t], false, false);
    }
  }

#pragma unroll
  for (int t = 0; t < NT; ++t) {
    const int n = n0 + 16 * t + idx;
    float bn = bias[n];
#pragma unroll
    for (int r = 0; r < 8; ++r) {
      float v = acc[t][r] + bn;
      v = (v > 20.f) ? v : log1pf(__expf(v));
      Cd[(size_t)(m0 + r + grp * 8) * N + n] = v;
    }
  }
}

// ------- final GEMM with TDM-staged B panel in LDS (out = y @ W_out^T) -------
// Block = 4 waves -> 64M x 64N; per 64-wide K chunk, wave 0 issues one
// tensor_load_to_lds for the 64x64 B panel (8 KB), waits TENSORcnt, workgroup
// barrier, then all waves read fragments from LDS (ds_load_b128): one TDM op
// feeds 32 WMMAs.
__global__ void __launch_bounds__(128)
wmma_gemm_nt_tdm_kernel(const _Float16* __restrict__ A,
                        const _Float16* __restrict__ Bw,
                        float* __restrict__ C,
                        int M, int N, int K)
{
  __shared__ _Float16 ldsB[64 * 64];   // 8 KB

  const int lane = threadIdx.x & 31;
  const int wave = threadIdx.x >> 5;
  const int grp  = lane >> 4;
  const int idx  = lane & 15;
  const int m0   = blockIdx.x * 64 + wave * 16;
  const int n0   = blockIdx.y * 64;

  const _Float16* arow = A + (size_t)(m0 + idx) * K;

  v8f acc[4] = {};
  for (int kc = 0; kc < K; kc += 64) {
#if HAVE_TDM
    if (threadIdx.x < 32) {   // one wave issues the DMA
      tdm_load_2d_f16((unsigned)(size_t)(void*)ldsB,
                      Bw + (size_t)n0 * K + kc,
                      (unsigned)N, (unsigned)K, 64u, 64u);
      __builtin_amdgcn_s_wait_tensorcnt(0);
    }
#else
    for (int i = threadIdx.x; i < 64 * 8; i += 128) {
      int r = i >> 3, c = i & 7;
      ((h8*)ldsB)[r * 8 + c] =
          *(const h8*)(Bw + (size_t)(n0 + r) * K + kc + c * 8);
    }
#endif
    __syncthreads();
#pragma unroll
    for (int k0 = 0; k0 < 64; k0 += 32) {
      AFrag a;
      a.h[0] = *(const h8*)(arow + kc + k0 + grp * 8);
      a.h[1] = *(const h8*)(arow + kc + k0 + 16 + grp * 8);
#pragma unroll
      for (int t = 0; t < 4; ++t) {
        AFrag b;
        const _Float16* bsrc = ldsB + (t * 16 + idx) * 64 + k0 + grp * 8;
        b.h[0] = *(const h8*)(bsrc);
        b.h[1] = *(const h8*)(bsrc + 16);
        acc[t] = __builtin_amdgcn_wmma_f32_16x16x32_f16(
            false, a.v, false, b.v, (short)0, acc[t], false, false);
      }
    }
    __syncthreads();
  }

#pragma unroll
  for (int t = 0; t < 4; ++t) {
    const int n = n0 + 16 * t + idx;
#pragma unroll
    for (int r = 0; r < 8; ++r)
      C[(size_t)(m0 + r + grp * 8) * N + n] = acc[t][r];
  }
}

// ---------------- causal depthwise conv (K=4) + silu, both dirs ----------------
__global__ void conv_silu_kernel(const float* __restrict__ xz,      // (BL, 2*ED)
                                 const float* __restrict__ w0,      // (ED, KCONV)
                                 const float* __restrict__ w1,
                                 const float* __restrict__ bias0,   // (ED,)
                                 const float* __restrict__ bias1,
                                 float* __restrict__ xconv,         // (2, BL, ED)
                                 _Float16* __restrict__ hxconv)     // (2, BL, ED)
{
  int i = blockIdx.x * blockDim.x + threadIdx.x;
  if (i >= BL * ED) return;
  const int dir = blockIdx.y;
  const float* w    = dir ? w1 : w0;
  const float* bias = dir ? bias1 : bias0;

  int e   = i % ED;
  int row = i / ED;          // branch-local b*L + l
  int b   = row / LSEQ;
  int l   = row % LSEQ;
  float acc = bias[e];
#pragma unroll
  for (int j = 0; j < KCONV; ++j) {
    int li = l + j - (KCONV - 1);
    if (li >= 0) {
      int t = dir ? (LSEQ - 1 - li) : li;
      acc += w[e * KCONV + j] * xz[((size_t)(b * LSEQ + t)) * (2 * ED) + e];
    }
  }
  float s = acc / (1.f + __expf(-acc));   // silu
  size_t o = (size_t)dir * BL * ED + i;
  xconv[o]  = s;
  hxconv[o] = (_Float16)s;
}

// ---------------- selective scan, both dirs ----------------
// One lane per (dir, b, e, n): 1 v_exp per step, cross-lane 16-way reduction
// via __shfl_xor (wave32 lane permute).  2*4*2048*16 = 262144 lanes = 8192 waves.
__global__ void scan_kernel(const float* __restrict__ delta,   // (2, BL, ED)
                            const float* __restrict__ dbc,     // (2, BL, 96)
                            const float* __restrict__ xconv,   // (2, BL, ED)
                            const float* __restrict__ xz,      // (BL, 2*ED)
                            const float* __restrict__ A_log0,  // (ED, 16)
                            const float* __restrict__ A_log1,
                            const float* __restrict__ Dp0,     // (ED,)
                            const float* __restrict__ Dp1,
                            float* __restrict__ y)             // (2, BL, ED)
{
  int tid  = blockIdx.x * blockDim.x + threadIdx.x;
  int n    = tid & (NSTATE - 1);
  int pair = tid >> 4;                       // dir*B_SZ*ED + b*ED + e
  if (pair >= 2 * B_SZ * ED) return;
  int dir = pair / (B_SZ * ED);
  int rem = pair - dir * (B_SZ * ED);
  int b   = rem / ED;
  int e   = rem - b * ED;

  const float* A_log = dir ? A_log1 : A_log0;
  float dpe = (dir ? Dp1 : Dp0)[e];
  float a   = -__expf(A_log[e * NSTATE + n]);
  float h   = 0.f;

  const size_t dbase = (size_t)dir * BL;
  for (int l = 0; l < LSEQ; ++l) {
    size_t row = dbase + (size_t)b * LSEQ + l;
    float dv = delta[row * ED + e];          // uniform across the 16 lanes
    float xv = xconv[row * ED + e];
    float Bn = dbc[row * 96 + 64 + n];
    float Cn = dbc[row * 96 + 80 + n];

    h = __expf(dv * a) * h + (dv * xv) * Bn;
    float p = h * Cn;
    p += __shfl_xor(p, 1, 32);
    p += __shfl_xor(p, 2, 32);
    p += __shfl_xor(p, 4, 32);
    p += __shfl_xor(p, 8, 32);

    if (n == 0) {
      int t = dir ? (LSEQ - 1 - l) : l;
      float zr = xz[((size_t)b * LSEQ + t) * (2 * ED) + ED + e];
      float zs = zr / (1.f + __expf(-zr));   // silu(z)
      y[row * ED + e] = (p + dpe * xv) * zs;
    }
  }
}

// ---------------- combine forward + reversed backward ----------------
__global__ void combine_kernel(const float* __restrict__ y,   // (2, BL, ED)
                               _Float16* __restrict__ ysum)   // (BL, ED)
{
  int i = blockIdx.x * blockDim.x + threadIdx.x;
  if (i >= BL * ED) return;
  int e   = i % ED;
  int row = i / ED;
  int b   = row / LSEQ;
  int l   = row % LSEQ;
  float v = 0.5f * (y[i] +
                    y[(size_t)BL * ED +
                      ((size_t)b * LSEQ + (LSEQ - 1 - l)) * ED + e]);
  ysum[i] = (_Float16)v;
}

// ---------------- launcher ----------------
extern "C" void kernel_launch(void* const* d_in, const int* in_sizes, int n_in,
                              void* d_out, int out_size, void* d_ws, size_t ws_size,
                              hipStream_t stream)
{
  const float* x       = (const float*)d_in[0];
  const float* W_in    = (const float*)d_in[1];
  const float* conv_w[2] = {(const float*)d_in[2],  (const float*)d_in[9]};
  const float* conv_b[2] = {(const float*)d_in[3],  (const float*)d_in[10]};
  const float* Wx[2]     = {(const float*)d_in[4],  (const float*)d_in[11]};
  const float* Wdt[2]    = {(const float*)d_in[5],  (const float*)d_in[12]};
  const float* b_dt[2]   = {(const float*)d_in[6],  (const float*)d_in[13]};
  const float* A_log[2]  = {(const float*)d_in[7],  (const float*)d_in[14]};
  const float* Dp[2]     = {(const float*)d_in[8],  (const float*)d_in[15]};
  const float* W_out   = (const float*)d_in[16];
  float* out = (float*)d_out;

  char* ws = (char*)d_ws;
  size_t off = 0;
  auto alloc = [&](size_t bytes) -> void* {
    void* p = ws + off;
    off += (bytes + 255) & ~(size_t)255;
    return p;
  };

  _Float16* hx      = (_Float16*)alloc((size_t)BL * D_MODEL * 2);       // 4 MB
  _Float16* hW_in   = (_Float16*)alloc((size_t)2 * ED * D_MODEL * 2);   // 8 MB
  _Float16* hW_out  = (_Float16*)alloc((size_t)D_MODEL * ED * 2);       // 4 MB
  _Float16* hWx[2]  = {(_Float16*)alloc((size_t)96 * ED * 2),
                       (_Float16*)alloc((size_t)96 * ED * 2)};
  _Float16* hWdt[2] = {(_Float16*)alloc((size_t)ED * DT_RANK * 2),
                       (_Float16*)alloc((size_t)ED * DT_RANK * 2)};
  float* xz      = (float*)alloc((size_t)BL * 2 * ED * 4);              // 32 MB
  float* xconv   = (float*)alloc((size_t)2 * BL * ED * 4);              // 32 MB
  _Float16* hxconv = (_Float16*)alloc((size_t)2 * BL * ED * 2);         // 16 MB
  float* dbc     = (float*)alloc((size_t)2 * BL * 96 * 4);              // 1.5 MB
  float* delta   = (float*)alloc((size_t)2 * BL * ED * 4);              // 32 MB
  float* y       = (float*)alloc((size_t)2 * BL * ED * 4);              // 32 MB
  _Float16* ysum = (_Float16*)alloc((size_t)BL * ED * 2);               // 8 MB

  const int TB = 256;
  auto cvt = [&](const float* s, _Float16* d, int n) {
    f32_to_f16_kernel<<<(n + TB - 1) / TB, TB, 0, stream>>>(s, d, n);
  };

  // 1) f16 copies of activations + weights
  cvt(x, hx, BL * D_MODEL);
  cvt(W_in, hW_in, 2 * ED * D_MODEL);
  cvt(W_out, hW_out, D_MODEL * ED);
  for (int d = 0; d < 2; ++d) {
    cvt(Wx[d], hWx[d], 96 * ED);
    cvt(Wdt[d], hWdt[d], ED * DT_RANK);
  }

  // 2) xz = x @ W_in^T   (2048 x 4096, K=1024), 16x64 strips
  wmma_gemm_nt_kernel<4, 0, 1><<<dim3(BL / 16, (2 * ED) / 64), 32, 0, stream>>>(
      hx, hW_in, nullptr, xz, BL, 2 * ED, D_MODEL, nullptr, nullptr);

  // 3) both branches fused per stage
  conv_silu_kernel<<<dim3((BL * ED + TB - 1) / TB, 2), TB, 0, stream>>>(
      xz, conv_w[0], conv_w[1], conv_b[0], conv_b[1], xconv, hxconv);

  // dbc = xconv @ Wx^T  (2 x 2048 x 96, K=2048), 16x32 strips
  wmma_gemm_nt_kernel<2, 0, 2><<<dim3(BL / 16, 96 / 32, 2), 32, 0, stream>>>(
      hxconv, hWx[0], hWx[1], dbc, BL, 96, ED, nullptr, nullptr);

  // delta = softplus(dbc[:, :64] @ Wdt^T + b_dt)  (2 x 2048 x 2048, K=64)
  wmma_gemm_nt_a32_kernel<4, 2><<<dim3(BL / 16, ED / 64, 2), 32, 0, stream>>>(
      dbc, hWdt[0], hWdt[1], delta, BL, ED, DT_RANK, 96, b_dt[0], b_dt[1]);

  // selective scan, both dirs
  scan_kernel<<<(2 * B_SZ * ED * NSTATE + TB - 1) / TB, TB, 0, stream>>>(
      delta, dbc, xconv, xz, A_log[0], A_log[1], Dp[0], Dp[1], y);

  // 4) y = 0.5 * (y_f + reverse(y_b)) -> f16
  combine_kernel<<<(BL * ED + TB - 1) / TB, TB, 0, stream>>>(y, ysum);

  // 5) out = y @ W_out^T  (2048 x 1024, K=2048), TDM-staged B panel
  wmma_gemm_nt_tdm_kernel<<<dim3(BL / 64, D_MODEL / 64), 128, 0, stream>>>(
      ysum, hW_out, out, BL, D_MODEL, ED);
}
